// myModel_45191645889034
// MI455X (gfx1250) — matrix operand order repaired and checked by reading, not compiled, not run on hardware
//
#include <hip/hip_runtime.h>

// ---------------------------------------------------------------------------
// Model dims (fixed by the reference)
// ---------------------------------------------------------------------------
#define N_LAYERS 4
#define EMB      256
#define SEQ      2048
#define BATCH    8
#define NROWS    (BATCH * SEQ)   // 16384

typedef __attribute__((ext_vector_type(16))) __bf16 bf16x16;
typedef __attribute__((ext_vector_type(8)))  __bf16 bf16x8;
typedef __attribute__((ext_vector_type(8)))  float  f32x8;

static __device__ __forceinline__ int lane_id() { return (int)(threadIdx.x & 31u); }

// ---------------------------------------------------------------------------
// Fragment loaders from row-major bf16 matrices (row stride ld, elements).
//
// A-fragment (16x32, M x K): lane L<16 holds row L, K = {0..7, 16..23};
// lane L+16 holds row L, K = {8..15, 24..31}; element order v0..v7 = first
// 8 K's pairwise, v8..v15 (VGPR4-7) = second 8 K's.
// ---------------------------------------------------------------------------
static __device__ __forceinline__ bf16x16
load_frag_a(const __bf16* __restrict__ src, int ld, int row0, int k0) {
  const int lane = lane_id();
  const int r = lane & 15, half = lane >> 4;
  const __bf16* p = src + (size_t)(row0 + r) * ld + k0 + half * 8;
  bf16x8 lo = *(const bf16x8*)(p);
  bf16x8 hi = *(const bf16x8*)(p + 16);
  bf16x16 out;
#pragma unroll
  for (int i = 0; i < 8; ++i) { out[i] = lo[i]; out[i + 8] = hi[i]; }
  return out;
}

// B-fragment (32x16, K x N): lane L<16 holds column L with K={0..7, 16..23};
// lane L+16 holds column L with K={8..15, 24..31}.  Source rows are the N-columns
// (e.g. rows of a row-major weight W[f,e] for X@W^T).
static __device__ __forceinline__ bf16x16
load_frag_b(const __bf16* __restrict__ src, int ld, int n0, int k0) {
  const int lane = lane_id();
  const int r = lane & 15, half = lane >> 4;
  const __bf16* p = src + (size_t)(n0 + r) * ld + k0 + half * 8;
  bf16x8 lo = *(const bf16x8*)(p);
  bf16x8 hi = *(const bf16x8*)(p + 16);
  bf16x16 out;
#pragma unroll
  for (int i = 0; i < 8; ++i) { out[i] = lo[i]; out[i + 8] = hi[i]; }
  return out;
}

static __device__ __forceinline__ f32x8 wmma_bf16(bf16x16 a, bf16x16 b, f32x8 c) {
  // emits v_wmma_f32_16x16x32_bf16
  return __builtin_amdgcn_wmma_f32_16x16x32_bf16(false, a, false, b, (short)0, c,
                                                 false, false);
}

// Accumulator layout: VGPR r, lane l -> (m = r + 8*(l>>4), n = l&15)

// ---------------------------------------------------------------------------
// Weight f32 -> bf16 conversion
// ---------------------------------------------------------------------------
__global__ void cvt_kernel(const float* __restrict__ in, __bf16* __restrict__ out,
                           int nelem) {
  int i = blockIdx.x * blockDim.x + threadIdx.x;
  if (i < nelem) out[i] = (__bf16)in[i];
}

// ---------------------------------------------------------------------------
// h = emb[x] + pos   -> bf16 [NROWS, EMB]
// ---------------------------------------------------------------------------
__global__ void embed_kernel(const int* __restrict__ x, const float* __restrict__ emb,
                             const float* __restrict__ pos, __bf16* __restrict__ H) {
  const int bs = blockIdx.x;          // b*SEQ + s
  const int e  = threadIdx.x;         // 0..255
  const int s  = bs & (SEQ - 1);
  const int idx = x[bs];
  H[(size_t)bs * EMB + e] = (__bf16)(emb[(size_t)idx * EMB + e] +
                                     pos[(size_t)s * EMB + e]);
}

// ---------------------------------------------------------------------------
// Y = relu(X @ W^T).  X: [NROWS, EMB] bf16 row-major. W: [EMB, EMB] bf16.
// Optionally stores row-major (Yrm) and/or per-batch transposed (Yt, [E x S]).
// Block = 256 threads (8 waves); wave tile = 16 rows x 64 cols.
// grid = (NROWS/128, EMB/64)
// ---------------------------------------------------------------------------
__global__ void proj_kernel(const __bf16* __restrict__ X, const __bf16* __restrict__ W,
                            __bf16* __restrict__ Yrm, __bf16* __restrict__ Yt) {
  const int w    = (int)(threadIdx.x >> 5);
  const int row0 = blockIdx.x * 128 + w * 16;
  const int col0 = blockIdx.y * 64;
  const int lane = lane_id();
  const int n = lane & 15, half = lane >> 4;

  bf16x16 afr[8];
#pragma unroll
  for (int k = 0; k < 8; ++k) afr[k] = load_frag_a(X, EMB, row0, k * 32);

#pragma unroll
  for (int nt = 0; nt < 4; ++nt) {
    f32x8 acc = {0.f, 0.f, 0.f, 0.f, 0.f, 0.f, 0.f, 0.f};
#pragma unroll
    for (int k = 0; k < 8; ++k) {
      bf16x16 bfr = load_frag_b(W, EMB, col0 + nt * 16, k * 32);
      acc = wmma_bf16(afr[k], bfr, acc);
    }
    if (Yrm) {
#pragma unroll
      for (int r = 0; r < 8; ++r) {
        float v = acc[r]; v = v > 0.f ? v : 0.f;
        Yrm[(size_t)(row0 + r + 8 * half) * EMB + col0 + nt * 16 + n] = (__bf16)v;
      }
    }
    if (Yt) {  // transposed per batch: Yt[b][f][s]; contiguous along m
      const int b = row0 / SEQ, sl = row0 % SEQ;
      bf16x8 pk;
#pragma unroll
      for (int r = 0; r < 8; ++r) { float v = acc[r]; pk[r] = (__bf16)(v > 0.f ? v : 0.f); }
      *(bf16x8*)(Yt + (size_t)b * EMB * SEQ + (size_t)(col0 + nt * 16 + n) * SEQ +
                 sl + 8 * half) = pk;
    }
  }
}

// ---------------------------------------------------------------------------
// Fused attention-like op:  O[b] = (A[b] @ B[b]^T) @ C[b]   (no softmax)
// A,B row-major bf16 [S,E] per batch; C supplied transposed Ct [E,S] per batch.
// Block = 256 threads (8 waves); waves pair up: pair p = w>>1 owns an s-tile
// of 16 rows, each wave of the pair owns half of E (128 cols) and computes the
// 16x32 score block itself (duplicated: cheap vs. a cross-wave barrier).
// grid = (SEQ/64, BATCH)
// ---------------------------------------------------------------------------
__global__ void attn_kernel(const __bf16* __restrict__ Aact,
                            const __bf16* __restrict__ Bact,
                            const __bf16* __restrict__ Ct,
                            __bf16* __restrict__ O) {
  __shared__ __align__(16) __bf16 sc_all[8][16 * 32];  // per-wave score staging
  const int w    = (int)(threadIdx.x >> 5);
  const int lane = lane_id();
  const int n = lane & 15, half = lane >> 4;
  const int b  = blockIdx.y;
  const int s0 = blockIdx.x * 64 + (w >> 1) * 16;
  const int e0 = (w & 1) * 128;

  const __bf16* Ab = Aact + (size_t)b * SEQ * EMB;
  const __bf16* Bb = Bact + (size_t)b * SEQ * EMB;
  const __bf16* Cb = Ct   + (size_t)b * EMB * SEQ;
  __bf16* sc = sc_all[w];

  bf16x16 afr[8];
#pragma unroll
  for (int k = 0; k < 8; ++k) afr[k] = load_frag_a(Ab, EMB, s0, k * 32);

  f32x8 oacc[8];
  const f32x8 fz = {0.f, 0.f, 0.f, 0.f, 0.f, 0.f, 0.f, 0.f};
#pragma unroll
  for (int i = 0; i < 8; ++i) oacc[i] = fz;

  for (int t0 = 0; t0 < SEQ; t0 += 32) {
    // ---- scores: two 16x16 tiles over t-columns [t0,t0+16) and [t0+16,t0+32)
    f32x8 sa = fz, sb = fz;
#pragma unroll
    for (int k = 0; k < 8; ++k) {
      bf16x16 b0 = load_frag_b(Bb, EMB, t0,      k * 32);
      bf16x16 b1 = load_frag_b(Bb, EMB, t0 + 16, k * 32);
      sa = wmma_bf16(afr[k], b0, sa);
      sb = wmma_bf16(afr[k], b1, sb);
    }
    // ---- accumulator layout -> A-fragment layout via per-wave LDS tile
#pragma unroll
    for (int r = 0; r < 8; ++r) {
      sc[(r + 8 * half) * 32 + n]      = (__bf16)sa[r];
      sc[(r + 8 * half) * 32 + 16 + n] = (__bf16)sb[r];
    }
    asm volatile("s_wait_dscnt 0" ::: "memory");  // in-wave cross-lane RAW
    bf16x16 sfr;
    {
      const __bf16* p = sc + (lane & 15) * 32 + half * 8;
      bf16x8 lo = *(const bf16x8*)p;
      bf16x8 hi = *(const bf16x8*)(p + 16);
#pragma unroll
      for (int i = 0; i < 8; ++i) { sfr[i] = lo[i]; sfr[i + 8] = hi[i]; }
    }
    // ---- out += score @ C  (B-fragments come from rows of Ct = columns of C)
#pragma unroll
    for (int et = 0; et < 8; ++et) {
      bf16x16 cfr = load_frag_b(Cb, SEQ, e0 + et * 16, t0);
      oacc[et] = wmma_bf16(sfr, cfr, oacc[et]);
    }
  }

  // ---- store O row-major bf16 (no relu here; relu happens after Wd proj)
#pragma unroll
  for (int et = 0; et < 8; ++et) {
#pragma unroll
    for (int r = 0; r < 8; ++r) {
      O[(size_t)b * SEQ * EMB + (size_t)(s0 + r + 8 * half) * EMB + e0 + et * 16 + n] =
          (__bf16)oacc[et][r];
    }
  }
}

// ---------------------------------------------------------------------------
// Y = X @ Wout^T + bout  -> f32 [NROWS, 256]
// grid = (NROWS/128, 4)
// ---------------------------------------------------------------------------
__global__ void final_kernel(const __bf16* __restrict__ X, const __bf16* __restrict__ W,
                             const float* __restrict__ bias, float* __restrict__ Y) {
  const int w    = (int)(threadIdx.x >> 5);
  const int row0 = blockIdx.x * 128 + w * 16;
  const int col0 = blockIdx.y * 64;
  const int lane = lane_id();
  const int n = lane & 15, half = lane >> 4;

  bf16x16 afr[8];
#pragma unroll
  for (int k = 0; k < 8; ++k) afr[k] = load_frag_a(X, EMB, row0, k * 32);

#pragma unroll
  for (int nt = 0; nt < 4; ++nt) {
    f32x8 acc = {0.f, 0.f, 0.f, 0.f, 0.f, 0.f, 0.f, 0.f};
#pragma unroll
    for (int k = 0; k < 8; ++k) {
      bf16x16 bfr = load_frag_b(W, EMB, col0 + nt * 16, k * 32);
      acc = wmma_bf16(afr[k], bfr, acc);
    }
    const float bv = bias[col0 + nt * 16 + n];
#pragma unroll
    for (int r = 0; r < 8; ++r) {
      Y[(size_t)(row0 + r + 8 * half) * EMB + col0 + nt * 16 + n] = acc[r] + bv;
    }
  }
}

// ---------------------------------------------------------------------------
// Host-side orchestration
// ---------------------------------------------------------------------------
extern "C" void kernel_launch(void* const* d_in, const int* in_sizes, int n_in,
                              void* d_out, int out_size, void* d_ws, size_t ws_size,
                              hipStream_t stream) {
  (void)in_sizes; (void)n_in; (void)out_size; (void)ws_size;

  const int*   x    = (const int*)d_in[0];
  const float* emb  = (const float*)d_in[1];
  const float* pos  = (const float*)d_in[2];
  const float* Wa   = (const float*)d_in[3];
  const float* Wb   = (const float*)d_in[4];
  const float* Wc   = (const float*)d_in[5];
  const float* Wd   = (const float*)d_in[6];
  const float* Wout = (const float*)d_in[7];
  const float* bout = (const float*)d_in[8];
  float* out = (float*)d_out;

  // Workspace layout (bf16 elements). Total ~44.2 MB.
  __bf16* WSB = (__bf16*)d_ws;
  const size_t WMAT = (size_t)EMB * EMB;         // 65536
  const size_t NE   = (size_t)NROWS * EMB;       // 4194304
  __bf16* wa   = WSB;                 // 4 * WMAT
  __bf16* wb   = wa + 4 * WMAT;
  __bf16* wc   = wb + 4 * WMAT;
  __bf16* wd   = wc + 4 * WMAT;
  __bf16* wo   = wd + 4 * WMAT;       // 1 * WMAT
  __bf16* H    = wo + WMAT;           // [NROWS, EMB]
  __bf16* Aact = H    + NE;           // [NROWS, EMB]
  __bf16* Bact = Aact + NE;           // [NROWS, EMB]
  __bf16* Ctr  = Bact + NE;           // [BATCH][EMB][SEQ]
  __bf16* Oact = Ctr  + NE;           // [NROWS, EMB]

  // Weights -> bf16
  {
    const int n4 = (int)(4 * WMAT);
    cvt_kernel<<<(n4 + 255) / 256, 256, 0, stream>>>(Wa, wa, n4);
    cvt_kernel<<<(n4 + 255) / 256, 256, 0, stream>>>(Wb, wb, n4);
    cvt_kernel<<<(n4 + 255) / 256, 256, 0, stream>>>(Wc, wc, n4);
    cvt_kernel<<<(n4 + 255) / 256, 256, 0, stream>>>(Wd, wd, n4);
    cvt_kernel<<<((int)WMAT + 255) / 256, 256, 0, stream>>>(Wout, wo, (int)WMAT);
  }

  embed_kernel<<<NROWS, EMB, 0, stream>>>(x, emb, pos, H);

  dim3 pgrid(NROWS / 128, EMB / 64);   // (128, 4)
  dim3 agrid(SEQ / 64, BATCH);         // (32, 8)
  for (int l = 0; l < N_LAYERS; ++l) {
    proj_kernel<<<pgrid, 256, 0, stream>>>(H, wa + (size_t)l * WMAT, Aact, nullptr);
    proj_kernel<<<pgrid, 256, 0, stream>>>(H, wb + (size_t)l * WMAT, Bact, nullptr);
    proj_kernel<<<pgrid, 256, 0, stream>>>(H, wc + (size_t)l * WMAT, nullptr, Ctr);
    attn_kernel<<<agrid, 256, 0, stream>>>(Aact, Bact, Ctr, Oact);
    proj_kernel<<<pgrid, 256, 0, stream>>>(Oact, wd + (size_t)l * WMAT, H, nullptr);
  }
  final_kernel<<<pgrid, 256, 0, stream>>>(H, wo, bout, out);
}
